// LabelScoringGraphModel_44040594653829
// MI455X (gfx1250) — matrix operand order repaired and checked by reading, not compiled, not run on hardware
//
#include <hip/hip_runtime.h>
#include <hip/hip_bf16.h>
#include <math.h>

// ---------------------------------------------------------------------------
// CDNA5 (gfx1250) implementation: BiLSTM -> label cosine scoring -> 2x
// GatedGraphConv(GRU) -> projection.  All GEMMs run through
// v_wmma_f32_16x16x32_bf16 (wave32, 16x16 tiles, f32 accumulate).
// B matrices are packed k-block-tiled: B[k/32][n][k%32] so both the
// global->LDS stage and the per-lane fragment reads are contiguous b128 ops.
// ---------------------------------------------------------------------------

typedef __attribute__((ext_vector_type(16))) __bf16 v16bf;
typedef __attribute__((ext_vector_type(8)))  float  v8f;

static constexpr int BATCH = 2048;
static constexpr int TSEQ  = 64;
static constexpr int EMB   = 300;
static constexpr int EPAD  = 320;   // K padded (mult of 64)
static constexpr int G4    = 1200;  // 4*H
static constexpr int G4P   = 1216;  // N padded (mult of 64)
static constexpr int CDIM  = 1024;
static constexpr int G3    = 3072;  // 3*C

__device__ __forceinline__ float sigf(float x) { return 1.0f / (1.0f + expf(-x)); }

// Tiled-B index: element (k, n) of an [K, Npad] matrix.
__device__ __forceinline__ long bidx(int k, int n, int Npad) {
    return (long)(k >> 5) * ((long)Npad * 32) + (long)n * 32 + (k & 31);
}

// ---------------------------------------------------------------------------
// bf16 WMMA GEMM:  C[M,N](f32) = [A0 | A1][M,K0+K1](bf16) * B(tiled) + bias
// Block tile 256x64xK64, 256 threads = 8 wave32 waves stacked along M,
// wave tile 32x64 (2x4 fragments) -> 16 WMMAs per K-step.
// K0, K1 multiples of 64; M multiple of 256; N multiple of 64.
// ---------------------------------------------------------------------------
__global__ void __launch_bounds__(256)
gemm_bf16_kernel(const __bf16* __restrict__ A0, int lda0, int K0,
                 const __bf16* __restrict__ A1, int lda1, int K1,
                 const __bf16* __restrict__ Bm, int ldbN,
                 const float* __restrict__ bias,
                 float* __restrict__ Cm, int ldc)
{
    __shared__ __attribute__((aligned(16))) __bf16 sA[256 * 64];     // [row][k 0..63]
    __shared__ __attribute__((aligned(16))) __bf16 sB[2 * 64 * 32];  // [khalf][n][k%32]

    const int tid  = threadIdx.x;
    const int lane = tid & 31;
    const int wm   = (tid >> 5) * 32;      // wave M offset in block tile
    const int rowBase = blockIdx.y * 256;
    const int colBase = blockIdx.x * 64;
    const int Ktot = K0 + K1;

    v8f acc[2][4] = {};

    const int m   = lane & 15;
    const int hsq = lane >> 4;        // 0/1 -> selects 16B chunk within A frag row
    const int n   = lane & 15;
    const int khq = (lane >> 4) * 2;  // 0/2 -> uint4 offset (16 bf16) within B frag

    for (int kb = 0; kb < Ktot; kb += 64) {
        const __bf16* Ap; int lda, kloc;
        if (kb < K0) { Ap = A0; lda = lda0; kloc = kb; }
        else         { Ap = A1; lda = lda1; kloc = kb - K0; }

        // ---- stage A tile: 256x64 bf16, one full row per thread.
        //      Load all 8 quads first (single clause), then store to LDS. ----
        {
            const uint4* ag = (const uint4*)(Ap + (long)(rowBase + tid) * lda + kloc);
            uint4 t[8];
            #pragma unroll
            for (int i = 0; i < 8; ++i) t[i] = ag[i];
            uint4* as = (uint4*)(sA + tid * 64);
            #pragma unroll
            for (int i = 0; i < 8; ++i) as[i] = t[i];
        }
        // ---- stage B tile: two contiguous 4KB chunks (tiled layout) ----
        {
            const long kbi = (long)(kb >> 5);
            const __bf16* bg0 = Bm + kbi * ((long)ldbN * 32) + (long)colBase * 32;
            const __bf16* bg1 = bg0 + (long)ldbN * 32;
            uint4 b0 = ((const uint4*)bg0)[tid];
            uint4 b1 = ((const uint4*)bg1)[tid];
            ((uint4*)sB)[tid]       = b0;
            ((uint4*)sB)[256 + tid] = b1;
            if (kb + 64 < Ktot)  // hint next B chunk (global_prefetch_b8)
                __builtin_prefetch(bg1 + (long)ldbN * 32, 0, 1);
        }
        __syncthreads();

        #pragma unroll
        for (int kh2 = 0; kh2 < 2; ++kh2) {
            union FragU { uint4 q[2]; v16bf v; };
            FragU af[2], bf[4];
            // A fragment (ISA 7.12.2): lanes 0-15 M=lane K[0..7|16..23],
            // lanes 16-31 same M, K[8..15|24..31] -> two aligned 16B chunks.
            #pragma unroll
            for (int fi = 0; fi < 2; ++fi) {
                const uint4* pa = (const uint4*)(sA + (wm + fi * 16 + m) * 64 + kh2 * 32);
                af[fi].q[0] = pa[hsq];
                af[fi].q[1] = pa[2 + hsq];
            }
            // B fragment: lanes 0-15 K[0..7|16..23], lanes 16-31 K[8..15|24..31]
            // of column n -> two aligned 16B chunks in tiled layout.
            #pragma unroll
            for (int fj = 0; fj < 4; ++fj) {
                const uint4* pb = (const uint4*)(sB + (kh2 * 64 + fj * 16 + n) * 32);
                bf[fj].q[0] = pb[hsq];
                bf[fj].q[1] = pb[2 + hsq];
            }
            #pragma unroll
            for (int fi = 0; fi < 2; ++fi)
                #pragma unroll
                for (int fj = 0; fj < 4; ++fj)
                    acc[fi][fj] = __builtin_amdgcn_wmma_f32_16x16x32_bf16(
                        false, af[fi].v, false, bf[fj].v,
                        (short)0, acc[fi][fj], false, false);
        }
        __syncthreads();
    }

    // ---- epilogue: C/D layout -- VGPR r: row=(lane>>4)*8+r, col=lane&15 ----
    const int n16 = lane & 15;
    const int m8  = (lane >> 4) * 8;
    #pragma unroll
    for (int fi = 0; fi < 2; ++fi) {
        #pragma unroll
        for (int fj = 0; fj < 4; ++fj) {
            const int col = colBase + fj * 16 + n16;
            const float bv = bias ? bias[col] : 0.0f;
            #pragma unroll
            for (int r = 0; r < 8; ++r) {
                const int row = rowBase + wm + fi * 16 + m8 + r;
                Cm[(long)row * ldc + col] = acc[fi][fj][r] + bv;
            }
        }
    }
}

// ---------------------------------------------------------------------------
// Helper / packing / fusion kernels
// ---------------------------------------------------------------------------
__global__ void zero_u32_kernel(unsigned* p, long n) {
    long i = (long)blockIdx.x * blockDim.x + threadIdx.x;
    if (i < n) p[i] = 0u;
}

__global__ void f32_to_bf16_kernel(const float* in, __bf16* out, long n) {
    long i = (long)blockIdx.x * blockDim.x + threadIdx.x;
    if (i < n) out[i] = (__bf16)in[i];
}

// Pack LSTM weights tiled: rows 0..299 = Wih^T, 320..619 = Whh^T, rest 0.
__global__ void pack_lstm_w_kernel(const float* __restrict__ Wih, const float* __restrict__ Whh,
                                   const float* __restrict__ bih, const float* __restrict__ bhh,
                                   __bf16* __restrict__ Wp, float* __restrict__ bias)
{
    int k = blockIdx.x; // 0..639
    for (int nn = threadIdx.x; nn < G4P; nn += blockDim.x) {
        float v = 0.0f;
        if (nn < G4) {
            if (k < EMB)                          v = Wih[(long)nn * EMB + k];
            else if (k >= EPAD && k < EPAD + EMB) v = Whh[(long)nn * EMB + (k - EPAD)];
        }
        Wp[bidx(k, nn, G4P)] = (__bf16)v;
    }
    if (k == 0)
        for (int nn = threadIdx.x; nn < G4P; nn += blockDim.x)
            bias[nn] = (nn < G4) ? bih[nn] + bhh[nn] : 0.0f;
}

// Transpose-pack tiled: Wp(k,n) = W[n,k], W is [N,K] row-major. grid = K.
__global__ void pack_wT_kernel(const float* __restrict__ W, int N, int K,
                               __bf16* __restrict__ Wp)
{
    int k = blockIdx.x;
    for (int nn = threadIdx.x; nn < N; nn += blockDim.x)
        Wp[bidx(k, nn, N)] = (__bf16)W[(long)nn * K + k];
}

// Direct-pack tiled: Wp(k,n) = W[k,n], W is [K,N] row-major. grid = K.
__global__ void pack_w_kernel(const float* __restrict__ W, int N,
                              __bf16* __restrict__ Wp)
{
    int k = blockIdx.x;
    for (int nn = threadIdx.x; nn < N; nn += blockDim.x)
        Wp[bidx(k, nn, N)] = (__bf16)W[(long)k * N + nn];
}

// Gather one timestep of embeddings -> [B, EPAD] bf16 (zero-padded).
__global__ void gather_x_kernel(const int* __restrict__ x, const float* __restrict__ emb,
                                int t, __bf16* __restrict__ xb)
{
    int b = blockIdx.x;
    int tok = x[(long)b * TSEQ + t];
    const float* row = emb + (long)tok * EMB;
    for (int j = threadIdx.x; j < EPAD; j += blockDim.x)
        xb[(long)b * EPAD + j] = (j < EMB) ? (__bf16)row[j] : (__bf16)0.0f;
}

// LSTM gate fusion: gates in G[B,1216] (order i,f,g,o), update c, h; emit hs slice.
__global__ void lstm_fuse_kernel(const float* __restrict__ G, float* __restrict__ c,
                                 __bf16* __restrict__ h, __bf16* __restrict__ hs_t)
{
    int b = blockIdx.x;
    const float* g = G + (long)b * G4P;
    for (int j = threadIdx.x; j < EMB; j += blockDim.x) {
        float ig = sigf(g[j]);
        float fg = sigf(g[EMB + j]);
        float gg = tanhf(g[2 * EMB + j]);
        float og = sigf(g[3 * EMB + j]);
        float cn = fg * c[(long)b * EMB + j] + ig * gg;
        c[(long)b * EMB + j] = cn;
        float hn = og * tanhf(cn);
        h[(long)b * EPAD + j]   = (__bf16)hn;
        hs_t[(long)b * EMB + j] = (__bf16)hn;
    }
}

// Per-(b): s[b,d] = sum_t normalize([hf(t,b,:), hb(t,b,:)])[d]; bf16 [B,640].
__global__ void cos_sum_kernel(const __bf16* __restrict__ hsf, const __bf16* __restrict__ hsb,
                               __bf16* __restrict__ s)
{
    int b = blockIdx.x, tid = threadIdx.x;
    __shared__ float red[256];
    float a0 = 0.f, a1 = 0.f, a2 = 0.f;
    for (int t = 0; t < TSEQ; ++t) {
        const __bf16* pf = hsf + ((long)t * BATCH + b) * EMB;
        const __bf16* pb = hsb + ((long)t * BATCH + b) * EMB;
        int d1 = tid + 256, d2 = tid + 512;
        float v0 = (float)pf[tid];
        float v1 = (d1 < EMB) ? (float)pf[d1] : (float)pb[d1 - EMB];
        float v2 = (d2 < 2 * EMB) ? (float)pb[d2 - EMB] : 0.0f;
        red[tid] = v0 * v0 + v1 * v1 + v2 * v2;
        __syncthreads();
        for (int off = 128; off > 0; off >>= 1) {
            if (tid < off) red[tid] += red[tid + off];
            __syncthreads();
        }
        float inv = 1.0f / fmaxf(sqrtf(red[0]), 1e-8f);
        __syncthreads();
        a0 += v0 * inv; a1 += v1 * inv;
        if (d2 < 2 * EMB) a2 += v2 * inv;
    }
    __bf16* so = s + (long)b * 640;
    so[tid]       = (__bf16)a0;
    so[tid + 256] = (__bf16)a1;
    if (tid + 512 < 600)      so[tid + 512] = (__bf16)a2;
    else if (tid + 512 < 640) so[tid + 512] = (__bf16)0.0f;
}

// Normalize label embeddings, pack transposed+tiled: lp(d,l). grid = C.
__global__ void label_pack_kernel(const float* __restrict__ L, __bf16* __restrict__ lp)
{
    int l = blockIdx.x, tid = threadIdx.x;
    __shared__ float red[256];
    float sq = 0.0f;
    for (int d = tid; d < 600; d += 256) { float v = L[(long)l * 600 + d]; sq += v * v; }
    red[tid] = sq;
    __syncthreads();
    for (int off = 128; off > 0; off >>= 1) {
        if (tid < off) red[tid] += red[tid + off];
        __syncthreads();
    }
    float inv = 1.0f / fmaxf(sqrtf(red[0]), 1e-8f);
    for (int d = tid; d < 640; d += 256) {
        float v = (d < 600) ? L[(long)l * 600 + d] * inv : 0.0f;
        lp[bidx(d, l, CDIM)] = (__bf16)v;
    }
}

// Edge scatter-add: out[dst[e], :] += m[src[e], :].  grid = NE.
__global__ void scatter_add_kernel(const int* __restrict__ ei, int NE,
                                   const float* __restrict__ m, float* __restrict__ out)
{
    long e = blockIdx.x;
    int sv = ei[e], dv = ei[(long)NE + e];
    const float* src = m + (long)sv * CDIM;
    float* dst = out + (long)dv * CDIM;
    for (int d = threadIdx.x; d < CDIM; d += blockDim.x)
        atomicAdd(&dst[d], src[d]);
}

// GRU fusion (torch GRUCell gate order r,z,n). Updates h (f32 + bf16 copies).
__global__ void gru_fuse_kernel(const float* __restrict__ gi, const float* __restrict__ gh,
                                float* __restrict__ hf, __bf16* __restrict__ hb)
{
    int b = blockIdx.x;
    long o = (long)b * G3;
    for (int j = threadIdx.x; j < CDIM; j += blockDim.x) {
        float r = sigf(gi[o + j] + gh[o + j]);
        float z = sigf(gi[o + CDIM + j] + gh[o + CDIM + j]);
        float n = tanhf(gi[o + 2 * CDIM + j] + r * gh[o + 2 * CDIM + j]);
        float hv = (1.0f - z) * n + z * hf[(long)b * CDIM + j];
        hf[(long)b * CDIM + j] = hv;
        hb[(long)b * CDIM + j] = (__bf16)hv;
    }
}

// ---------------------------------------------------------------------------
// Host orchestration
// ---------------------------------------------------------------------------
static inline void launch_gemm(const void* A0, int lda0, int K0,
                               const void* A1, int lda1, int K1,
                               const void* Bm, const float* bias,
                               float* C, int ldc, int M, int N, hipStream_t s)
{
    dim3 grid(N / 64, M / 256);
    gemm_bf16_kernel<<<grid, 256, 0, s>>>((const __bf16*)A0, lda0, K0,
                                          (const __bf16*)A1, lda1, K1,
                                          (const __bf16*)Bm, N, bias, C, ldc);
}

static inline void launch_zero(void* p, size_t bytes, hipStream_t s)
{
    long n = (long)(bytes / 4);
    zero_u32_kernel<<<dim3((unsigned)((n + 255) / 256)), 256, 0, s>>>((unsigned*)p, n);
}

extern "C" void kernel_launch(void* const* d_in, const int* in_sizes, int n_in,
                              void* d_out, int out_size, void* d_ws, size_t ws_size,
                              hipStream_t stream)
{
    const int*   x       = (const int*)d_in[0];
    const int*   ei      = (const int*)d_in[1];
    const float* emb     = (const float*)d_in[2];
    const float* Wih_f   = (const float*)d_in[3];
    const float* Whh_f   = (const float*)d_in[4];
    const float* bih_f   = (const float*)d_in[5];
    const float* bhh_f   = (const float*)d_in[6];
    const float* Wih_b   = (const float*)d_in[7];
    const float* Whh_b   = (const float*)d_in[8];
    const float* bih_b   = (const float*)d_in[9];
    const float* bhh_b   = (const float*)d_in[10];
    const float* labelE  = (const float*)d_in[11];
    const float* ggcW    = (const float*)d_in[12];
    const float* gruWih  = (const float*)d_in[13];
    const float* gruWhh  = (const float*)d_in[14];
    const float* gru_bih = (const float*)d_in[15];
    const float* gru_bhh = (const float*)d_in[16];
    const float* projW   = (const float*)d_in[17];
    const float* proj_b  = (const float*)d_in[18];
    float* out = (float*)d_out;
    const int NE = in_sizes[1] / 2;

    // -------- workspace carve-out --------
    char* ws = (char*)d_ws;
    auto alloc = [&](size_t bytes) -> char* {
        char* p = ws;
        ws += (bytes + 255) & ~(size_t)255;
        return p;
    };
    __bf16* WpF    = (__bf16*)alloc((size_t)2 * EPAD * G4P * sizeof(__bf16)); // 640x1216
    __bf16* WpB    = (__bf16*)alloc((size_t)2 * EPAD * G4P * sizeof(__bf16));
    float*  biasF  = (float*) alloc((size_t)G4P * sizeof(float));
    float*  biasB  = (float*) alloc((size_t)G4P * sizeof(float));
    __bf16* xb     = (__bf16*)alloc((size_t)BATCH * EPAD * sizeof(__bf16));
    float*  Gbuf   = (float*) alloc((size_t)BATCH * G4P * sizeof(float));
    __bf16* hstate = (__bf16*)alloc((size_t)BATCH * EPAD * sizeof(__bf16));
    float*  cstate = (float*) alloc((size_t)BATCH * EMB * sizeof(float));
    __bf16* hsF    = (__bf16*)alloc((size_t)TSEQ * BATCH * EMB * sizeof(__bf16));
    __bf16* hsB    = (__bf16*)alloc((size_t)TSEQ * BATCH * EMB * sizeof(__bf16));
    __bf16* sbuf   = (__bf16*)alloc((size_t)BATCH * 640 * sizeof(__bf16));
    __bf16* lp     = (__bf16*)alloc((size_t)640 * CDIM * sizeof(__bf16));
    float*  h_f32  = (float*) alloc((size_t)BATCH * CDIM * sizeof(float));
    __bf16* h_bf16 = (__bf16*)alloc((size_t)BATCH * CDIM * sizeof(__bf16));
    __bf16* ggcWp  = (__bf16*)alloc((size_t)2 * CDIM * CDIM * sizeof(__bf16));
    __bf16* gWihP  = (__bf16*)alloc((size_t)CDIM * G3 * sizeof(__bf16));
    __bf16* gWhhP  = (__bf16*)alloc((size_t)CDIM * G3 * sizeof(__bf16));
    __bf16* projWp = (__bf16*)alloc((size_t)CDIM * CDIM * sizeof(__bf16));
    float*  m_f32  = (float*) alloc((size_t)BATCH * CDIM * sizeof(float));
    float*  magg   = (float*) alloc((size_t)BATCH * CDIM * sizeof(float));
    __bf16* m_bf16 = (__bf16*)alloc((size_t)BATCH * CDIM * sizeof(__bf16));
    float*  gi     = (float*) alloc((size_t)BATCH * G3 * sizeof(float));
    float*  gh     = (float*) alloc((size_t)BATCH * G3 * sizeof(float));
    (void)ws_size; (void)n_in; (void)out_size;

    // -------- weight packing (bf16, padded/transposed, k-block tiled) --------
    pack_lstm_w_kernel<<<2 * EPAD, 256, 0, stream>>>(Wih_f, Whh_f, bih_f, bhh_f, WpF, biasF);
    pack_lstm_w_kernel<<<2 * EPAD, 256, 0, stream>>>(Wih_b, Whh_b, bih_b, bhh_b, WpB, biasB);
    for (int layer = 0; layer < 2; ++layer)
        pack_w_kernel<<<CDIM, 256, 0, stream>>>(ggcW + (size_t)layer * CDIM * CDIM, CDIM,
                                                ggcWp + (size_t)layer * CDIM * CDIM);
    pack_wT_kernel<<<CDIM, 256, 0, stream>>>(gruWih, G3, CDIM, gWihP);
    pack_wT_kernel<<<CDIM, 256, 0, stream>>>(gruWhh, G3, CDIM, gWhhP);
    pack_wT_kernel<<<CDIM, 256, 0, stream>>>(projW, CDIM, CDIM, projWp);
    label_pack_kernel<<<CDIM, 256, 0, stream>>>(labelE, lp);

    // -------- forward LSTM --------
    launch_zero(cstate, (size_t)BATCH * EMB * sizeof(float), stream);
    launch_zero(hstate, (size_t)BATCH * EPAD * sizeof(__bf16), stream);
    for (int t = 0; t < TSEQ; ++t) {
        gather_x_kernel<<<BATCH, 256, 0, stream>>>(x, emb, t, xb);
        launch_gemm(xb, EPAD, EPAD, hstate, EPAD, EPAD, WpF, biasF,
                    Gbuf, G4P, BATCH, G4P, stream);
        lstm_fuse_kernel<<<BATCH, 256, 0, stream>>>(Gbuf, cstate, hstate,
                                                    hsF + (long)t * BATCH * EMB);
    }
    // -------- backward LSTM (time-aligned, scan from T-1 down) --------
    launch_zero(cstate, (size_t)BATCH * EMB * sizeof(float), stream);
    launch_zero(hstate, (size_t)BATCH * EPAD * sizeof(__bf16), stream);
    for (int tt = 0; tt < TSEQ; ++tt) {
        int t = TSEQ - 1 - tt;
        gather_x_kernel<<<BATCH, 256, 0, stream>>>(x, emb, t, xb);
        launch_gemm(xb, EPAD, EPAD, hstate, EPAD, EPAD, WpB, biasB,
                    Gbuf, G4P, BATCH, G4P, stream);
        lstm_fuse_kernel<<<BATCH, 256, 0, stream>>>(Gbuf, cstate, hstate,
                                                    hsB + (long)t * BATCH * EMB);
    }

    // -------- label cosine scoring: ls = (sum_t cn) @ ln^T --------
    cos_sum_kernel<<<BATCH, 256, 0, stream>>>(hsF, hsB, sbuf);
    launch_gemm(sbuf, 640, 640, nullptr, 0, 0, lp, nullptr,
                h_f32, CDIM, BATCH, CDIM, stream);
    {
        long n = (long)BATCH * CDIM;
        f32_to_bf16_kernel<<<dim3((unsigned)((n + 255) / 256)), 256, 0, stream>>>(h_f32, h_bf16, n);
    }

    // -------- GatedGraphConv x2 --------
    for (int layer = 0; layer < 2; ++layer) {
        launch_gemm(h_bf16, CDIM, CDIM, nullptr, 0, 0,
                    ggcWp + (size_t)layer * CDIM * CDIM, nullptr,
                    m_f32, CDIM, BATCH, CDIM, stream);
        launch_zero(magg, (size_t)BATCH * CDIM * sizeof(float), stream);
        scatter_add_kernel<<<NE, 256, 0, stream>>>(ei, NE, m_f32, magg);
        {
            long n = (long)BATCH * CDIM;
            f32_to_bf16_kernel<<<dim3((unsigned)((n + 255) / 256)), 256, 0, stream>>>(magg, m_bf16, n);
        }
        launch_gemm(m_bf16, CDIM, CDIM, nullptr, 0, 0, gWihP, gru_bih,
                    gi, G3, BATCH, G3, stream);
        launch_gemm(h_bf16, CDIM, CDIM, nullptr, 0, 0, gWhhP, gru_bhh,
                    gh, G3, BATCH, G3, stream);
        gru_fuse_kernel<<<BATCH, 256, 0, stream>>>(gi, gh, h_f32, h_bf16);
    }

    // -------- projection --------
    launch_gemm(h_bf16, CDIM, CDIM, nullptr, 0, 0, projWp, proj_b,
                out, CDIM, BATCH, CDIM, stream);
}
